// fengheNet_63539746177640
// MI455X (gfx1250) — compile-verified, matrix operand order checked by reading, not run on hardware
//
#include <hip/hip_runtime.h>
#include <hip/hip_fp16.h>

// ===========================================================================
// GRNet-style point completion forward for MI455X (gfx1250, wave32, WMMA).
// Dense GEMMs: v_wmma_f32_16x16x32_f16, BK=64 LDS tiles, double-buffered and
// staged with CDNA5 global_load_async_to_lds_b128 (ASYNCcnt) overlapped with
// WMMA compute; depth-2 L2 prefetch via global_prefetch_b8.
// 3D conv/deconv: implicit GEMM with on-the-fly A-tile gather.
// Point sampling approximated by the partial cloud (deterministic).
// ===========================================================================

typedef _Float16 h16;
typedef __attribute__((ext_vector_type(16))) _Float16 v16h;
typedef __attribute__((ext_vector_type(8)))  float    v8f;

#define LDS_K 40   // conv tile: 32 data + 8 pad halves per row
#define LDS_KB 72  // gemm tile: 64 data + 8 pad halves per row (144B, 16B-mult)

// Load one 16x32 f16 WMMA fragment from an LDS tile ([16 rows][ld halves]).
// ISA layout: lanes 0-15 = rows M0..15 with K {0..7,16..23}; lanes 16-31 same
// rows with K {8..15,24..31}.
static __device__ __forceinline__ v16h ld_frag(const h16* tile, int ld, int lane) {
  const h16* p = tile + (lane & 15) * ld + ((lane >> 4) << 3);
  union { v16h v; uint4 q[2]; } u;
  u.q[0] = *(const uint4*)p;
  u.q[1] = *(const uint4*)(p + 16);
  return u.v;
}

static __device__ __forceinline__ v8f wmma_f16(v16h a, v16h b, v8f c) {
  // (neg_a, A, neg_b, B, c_mod, C, reuse_a, reuse_b)
  return __builtin_amdgcn_wmma_f32_16x16x32_f16(false, a, false, b, (short)0, c,
                                                false, false);
}

// CDNA5 async copy: global -> LDS, 16 bytes per lane, tracked by ASYNCcnt.
static __device__ __forceinline__ void async_b128(const h16* gptr, h16* lptr) {
  unsigned lds = (unsigned)(unsigned long long)(void*)lptr;  // LDS byte offset
  asm volatile("global_load_async_to_lds_b128 %0, %1, off"
               :: "v"(lds), "v"(gptr) : "memory");
}
static __device__ __forceinline__ void wait_async0() {
  asm volatile("s_wait_asynccnt 0" ::: "memory");
}

// ---------------------------------------------------------------------------
// GEMM: C[Mp][Np](f16) = act(A[Mp][Kp](f16) * Bt[Np][Kp]^T + bias)
// Block = 256 threads = 8 waves; tile 128x64x64; wave computes 32x32;
// 8 WMMAs per LDS stage (2 K-steps); double-buffered async pipeline:
//   L2 prefetch (i+2)  ->  async-to-LDS (i+1)  ->  WMMA (i)
// ---------------------------------------------------------------------------
__global__ __launch_bounds__(256) void k_wmma_gemm(
    const h16* __restrict__ A, const h16* __restrict__ Bt,
    const float* __restrict__ bias, h16* __restrict__ C,
    long N, long Kp, long Np, int act)
{
  __shared__ __align__(16) h16 As[2][128 * LDS_KB];
  __shared__ __align__(16) h16 Bs[2][64 * LDS_KB];
  const int t = threadIdx.x, lane = t & 31, wid = t >> 5;
  const long bm = (long)blockIdx.y * 128;
  const long bn = (long)blockIdx.x * 64;
  const int wm = (wid & 3) * 32, wn = (wid >> 2) * 32;
  v8f acc[2][2] = {};

  auto stage = [&](long k0, int buf) {
    // A tile: 128 rows x 64 halves = 1024 16B segments, 4 per thread (async)
#pragma unroll
    for (int q = 0; q < 4; ++q) {
      int s = t + q * 256;
      int r = s >> 3, col = (s & 7) << 3;
      async_b128(A + (bm + r) * Kp + k0 + col, &As[buf][r * LDS_KB + col]);
    }
    // B tile: 64 rows x 64 halves = 512 16B segments, 2 per thread (async)
#pragma unroll
    for (int q = 0; q < 2; ++q) {
      int s = t + q * 256;
      int r = s >> 3, col = (s & 7) << 3;
      async_b128(Bt + (bn + r) * Kp + k0 + col, &Bs[buf][r * LDS_KB + col]);
    }
  };

  const int nt = (int)(Kp >> 6);
  stage(0, 0);
  for (int it = 0; it < nt; ++it) {
    wait_async0();    // my async stores for tile `it` have landed in LDS
    __syncthreads();  // everyone's tile `it` visible; buf[it&1] reads of it-2 done
    if (it + 1 < nt) {
      stage((long)(it + 1) << 6, (it + 1) & 1);  // overlaps with WMMAs below
      if (it + 2 < nt)  // depth-2 prefetch into L2 (global_prefetch_b8)
        __builtin_prefetch(A + (bm + (t >> 1)) * Kp + ((long)(it + 2) << 6), 0, 1);
    }
    const h16* Ab = As[it & 1];
    const h16* Bb = Bs[it & 1];
#pragma unroll
    for (int ks = 0; ks < 2; ++ks) {
      const int ko = ks << 5;
      v16h a0 = ld_frag(Ab + (long)wm * LDS_KB + ko, LDS_KB, lane);
      v16h a1 = ld_frag(Ab + (long)(wm + 16) * LDS_KB + ko, LDS_KB, lane);
      v16h b0 = ld_frag(Bb + (long)wn * LDS_KB + ko, LDS_KB, lane);
      v16h b1 = ld_frag(Bb + (long)(wn + 16) * LDS_KB + ko, LDS_KB, lane);
      acc[0][0] = wmma_f16(a0, b0, acc[0][0]);
      acc[0][1] = wmma_f16(a0, b1, acc[0][1]);
      acc[1][0] = wmma_f16(a1, b0, acc[1][0]);
      acc[1][1] = wmma_f16(a1, b1, acc[1][1]);
    }
  }
#pragma unroll
  for (int i = 0; i < 2; ++i)
#pragma unroll
    for (int j = 0; j < 2; ++j) {
      long nc = bn + wn + j * 16 + (lane & 15);
      float bv = (bias && nc < N) ? bias[nc] : 0.f;
      long mr0 = bm + wm + i * 16 + ((lane >> 4) << 3);
#pragma unroll
      for (int v = 0; v < 8; ++v) {
        float x = (nc < N) ? (acc[i][j][v] + bv) : 0.f;
        if (act == 1) x = fmaxf(x, 0.f);
        else if (act == 2) x = (x > 0.f) ? x : 0.2f * x;
        C[(mr0 + v) * Np + nc] = (h16)x;
      }
    }
}

// ---------------------------------------------------------------------------
// Implicit-GEMM 3D conv (4x4x4, pad 2, stride 1; dilated=1 => lhs_dilation=2
// transposed conv). A[vox][Cin*64] gathered on the fly from NCDHW f32 volume.
// Wt is pre-packed [Coutp][Kp] f16 (deconv taps pre-flipped). Raw f16 output
// (no bias: it cancels under the following BatchNorm).
// ---------------------------------------------------------------------------
__global__ __launch_bounds__(256) void k_wmma_conv(
    const float* __restrict__ X, const h16* __restrict__ Wt,
    h16* __restrict__ C, int Din, int Dout, int Cin, long Kp, long Np,
    int dilated)
{
  __shared__ __align__(16) h16 As[128 * LDS_K];
  __shared__ __align__(16) h16 Bs[64 * LDS_K];
  const int t = threadIdx.x, lane = t & 31, wid = t >> 5;
  const long bm = (long)blockIdx.y * 128;
  const long bn = (long)blockIdx.x * 64;
  const int wm = (wid & 3) * 32, wn = (wid >> 2) * 32;
  const int ar = t >> 1, ac = (t & 1) << 4;
  const int br = t >> 2, bc = (t & 3) << 3;
  const long Mvox = (long)Dout * Dout * Dout;
  const long Kreal = (long)Cin * 64;
  long vox = bm + ar;
  int od = 0, oh = 0, ow = 0;
  const bool vok = vox < Mvox;
  if (vok) {
    od = (int)(vox / ((long)Dout * Dout));
    long rem = vox - (long)od * Dout * Dout;
    oh = (int)(rem / Dout);
    ow = (int)(rem % Dout);
  }
  v8f acc[2][2] = {};
  for (long k0 = 0; k0 < Kp; k0 += 32) {
    __syncthreads();
    {
      h16* dst = As + ar * LDS_K + ac;
      for (int u = 0; u < 16; ++u) {
        long k = k0 + ac + u;
        float val = 0.f;
        if (vok && k < Kreal) {
          int ci = (int)(k >> 6), kk = (int)(k & 63);
          int vd = od + (kk >> 4) - 2;
          int vh = oh + ((kk >> 2) & 3) - 2;
          int vw = ow + (kk & 3) - 2;
          if (dilated) {
            int lim = 2 * Din - 1;
            if ((vd | vh | vw) >= 0 && vd < lim && vh < lim && vw < lim &&
                (((vd | vh | vw) & 1) == 0))
              val = X[(((long)ci * Din + (vd >> 1)) * Din + (vh >> 1)) * Din +
                      (vw >> 1)];
          } else {
            if ((vd | vh | vw) >= 0 && vd < Din && vh < Din && vw < Din)
              val = X[(((long)ci * Din + vd) * Din + vh) * Din + vw];
          }
        }
        dst[u] = (h16)val;
      }
      *(uint4*)(Bs + br * LDS_K + bc) =
          *(const uint4*)(Wt + (bn + br) * Kp + k0 + bc);
    }
    __syncthreads();
    v16h a0 = ld_frag(As + (long)wm * LDS_K, LDS_K, lane);
    v16h a1 = ld_frag(As + (long)(wm + 16) * LDS_K, LDS_K, lane);
    v16h b0 = ld_frag(Bs + (long)wn * LDS_K, LDS_K, lane);
    v16h b1 = ld_frag(Bs + (long)(wn + 16) * LDS_K, LDS_K, lane);
    acc[0][0] = wmma_f16(a0, b0, acc[0][0]);
    acc[0][1] = wmma_f16(a0, b1, acc[0][1]);
    acc[1][0] = wmma_f16(a1, b0, acc[1][0]);
    acc[1][1] = wmma_f16(a1, b1, acc[1][1]);
  }
#pragma unroll
  for (int i = 0; i < 2; ++i)
#pragma unroll
    for (int j = 0; j < 2; ++j) {
      long nc = bn + wn + j * 16 + (lane & 15);
      long mr0 = bm + wm + i * 16 + ((lane >> 4) << 3);
#pragma unroll
      for (int v = 0; v < 8; ++v)
        C[(mr0 + v) * Np + nc] = (h16)acc[i][j][v];
    }
}

// ------------------------------- aux kernels -------------------------------

// lin weight (K_in x N_out) f32 -> Bt [Np][Kp] f16 (transposed, zero padded)
__global__ void k_pack_lin(const float* __restrict__ W, h16* __restrict__ Bt,
                           long K, long N, long Kp, long Np)
{
  long i = (long)blockIdx.x * 256 + threadIdx.x;
  if (i >= Np * Kp) return;
  long n = i / Kp, k = i % Kp;
  Bt[i] = (h16)((n < N && k < K) ? W[k * N + n] : 0.f);
}

// conv weight OIDHW (O x I*64) -> Bt [Op][Kp]
__global__ void k_pack_convw(const float* __restrict__ W, h16* __restrict__ Bt,
                             int O, int I, long Kp, long Np)
{
  long i = (long)blockIdx.x * 256 + threadIdx.x;
  if (i >= Np * Kp) return;
  long n = i / Kp, k = i % Kp;
  long K = (long)I * 64;
  Bt[i] = (h16)((n < O && k < K) ? W[n * K + k] : 0.f);
}

// deconv weight IODHW, flipped taps: dst[o][i*64+t] = W[i][o][flip(t)]
__global__ void k_pack_deconvw(const float* __restrict__ W, h16* __restrict__ Bt,
                               int O, int I, long Kp, long Np)
{
  long i = (long)blockIdx.x * 256 + threadIdx.x;
  if (i >= Np * Kp) return;
  long o = i / Kp, k = i % Kp;
  long K = (long)I * 64;
  float v = 0.f;
  if (o < O && k < K) {
    int ci = (int)(k >> 6), tp = (int)(k & 63);
    int ft = (3 - (tp >> 4)) * 16 + (3 - ((tp >> 2) & 3)) * 4 + (3 - (tp & 3));
    v = W[((long)ci * O + o) * 64 + ft];
  }
  Bt[i] = (h16)v;
}

// per-channel sum / sumsq over conv output [M][Np] f16, channels-last
__global__ void k_bn_stats(const h16* __restrict__ Cm, long M, long Np, int C,
                           float* __restrict__ st)
{
  long i = (long)blockIdx.x * 256 + threadIdx.x;
  if (i >= M * C) return;
  long m = i / C; int c = (int)(i % C);
  float v = (float)Cm[m * Np + c];
  atomicAdd(&st[c], v);
  atomicAdd(&st[C + c], v * v);
}

// BN + leaky-relu(0.2) + 2x2x2 maxpool -> channels-first f32 volume
__global__ void k_bn_lrelu_pool(const h16* __restrict__ Cm,
                                const float* __restrict__ st, long Np, int C,
                                int Do, int Dp, float* __restrict__ vol)
{
  long i = (long)blockIdx.x * 256 + threadIdx.x;
  long pv = (long)Dp * Dp * Dp;
  if (i >= (long)C * pv) return;
  int c = (int)(i / pv);
  long rem = i % pv;
  int pd = (int)(rem / ((long)Dp * Dp));
  int ph = (int)((rem / Dp) % Dp);
  int pw = (int)(rem % Dp);
  float M = (float)Do * Do * Do;
  float mean = st[c] / M;
  float var = st[C + c] / M - mean * mean;
  float inv = rsqrtf(var + 1e-5f);
  float best = -1e30f;
  for (int dd = 0; dd < 2; ++dd)
    for (int hh = 0; hh < 2; ++hh)
      for (int ww = 0; ww < 2; ++ww) {
        long vox = ((long)(2 * pd + dd) * Do + (2 * ph + hh)) * Do + (2 * pw + ww);
        float x = ((float)Cm[vox * Np + c] - mean) * inv;
        x = (x > 0.f) ? x : 0.2f * x;
        best = fmaxf(best, x);
      }
  vol[i] = best;
}

// BN + relu + skip-add -> channels-first f32 volume (decoder)
__global__ void k_bn_relu_add(const h16* __restrict__ Cm,
                              const float* __restrict__ st, long Np, int C,
                              int Do, const float* __restrict__ skip,
                              float* __restrict__ vol)
{
  long M = (long)Do * Do * Do;
  long i = (long)blockIdx.x * 256 + threadIdx.x;
  if (i >= (long)C * M) return;
  int c = (int)(i / M);
  long vox = i % M;
  float mean = st[c] / (float)M;
  float var = st[C + c] / (float)M - mean * mean;
  float inv = rsqrtf(var + 1e-5f);
  float x = ((float)Cm[vox * Np + c] - mean) * inv;
  x = fmaxf(x, 0.f);
  vol[i] = x + skip[i];
}

__global__ void k_gridding(const float* __restrict__ pts, int n,
                           float* __restrict__ g)
{
  int p = blockIdx.x * 256 + threadIdx.x;
  if (p >= n) return;
  float fr[3]; int lo[3];
  for (int j = 0; j < 3; ++j) {
    float c = (pts[p * 3 + j] + 1.f) * 64.f;
    float fl = floorf(c);
    lo[j] = (int)fl; fr[j] = c - fl;
  }
  for (int dx = 0; dx < 2; ++dx)
    for (int dy = 0; dy < 2; ++dy)
      for (int dz = 0; dz < 2; ++dz) {
        float w = (dx ? fr[0] : 1.f - fr[0]) * (dy ? fr[1] : 1.f - fr[1]) *
                  (dz ? fr[2] : 1.f - fr[2]);
        int ix = lo[0] + dx; ix = ix < 0 ? 0 : (ix > 127 ? 127 : ix);
        int iy = lo[1] + dy; iy = iy < 0 ? 0 : (iy > 127 ? 127 : iy);
        int iz = lo[2] + dz; iz = iz < 0 ? 0 : (iz > 127 ? 127 : iz);
        atomicAdd(&g[((long)ix * 128 + iy) * 128 + iz], w);
      }
}

__global__ void k_grid_rev(const float* __restrict__ g, float* __restrict__ out)
{
  const int S = 128;
  long i = (long)blockIdx.x * 256 + threadIdx.x;
  if (i >= (long)S * S * S) return;
  int x = (int)(i / (S * S)), y = (int)((i / S) % S), z = (int)(i % S);
  float s = 0.f, wx = 0.f, wy = 0.f, wz = 0.f;
  for (int dx = 0; dx < 2; ++dx)
    for (int dy = 0; dy < 2; ++dy)
      for (int dz = 0; dz < 2; ++dz) {
        int ax = x + dx, ay = y + dy, az = z + dz;
        float v = (ax < S && ay < S && az < S) ? g[((long)ax * S + ay) * S + az]
                                               : 0.f;
        s += v;
        wx += v * ((float)ax / 64.f - 1.f);
        wy += v * ((float)ay / 64.f - 1.f);
        wz += v * ((float)az / 64.f - 1.f);
      }
  float inv = 1.f / fmaxf(s, 1e-8f);
  out[i * 3 + 0] = (s > 0.f) ? wx * inv : 0.f;
  out[i * 3 + 1] = (s > 0.f) ? wy * inv : 0.f;
  out[i * 3 + 2] = (s > 0.f) ? wz * inv : 0.f;
}

// trilinear-corner gather: writes 8*C f16 features per point into X columns
__global__ void k_cubic(const float* __restrict__ pts, int n,
                        const float* __restrict__ vol, int C, int S,
                        h16* __restrict__ X, long ldx, long col0)
{
  long i = (long)blockIdx.x * 256 + threadIdx.x;
  if (i >= (long)n * C) return;
  int p = (int)(i / C), c = (int)(i % C);
  float half = S * 0.5f;
  int lx = (int)floorf((pts[p * 3 + 0] + 1.f) * half);
  int ly = (int)floorf((pts[p * 3 + 1] + 1.f) * half);
  int lz = (int)floorf((pts[p * 3 + 2] + 1.f) * half);
  int corner = 0;
  for (int dx = 0; dx < 2; ++dx)
    for (int dy = 0; dy < 2; ++dy)
      for (int dz = 0; dz < 2; ++dz) {
        int ix = lx + dx; ix = ix < 0 ? 0 : (ix > S - 1 ? S - 1 : ix);
        int iy = ly + dy; iy = iy < 0 ? 0 : (iy > S - 1 ? S - 1 : iy);
        int iz = lz + dz; iz = iz < 0 ? 0 : (iz > S - 1 ? S - 1 : iz);
        X[(long)p * ldx + col0 + (long)corner * C + c] =
            (h16)vol[(long)c * S * S * S + ((long)ix * S + iy) * S + iz];
        corner++;
      }
}

// brute-force 32-NN with radius fallback (matches top_k(-d2) + radius mask)
__global__ void k_knn(const float* __restrict__ xyz, int n, int nq, float r2,
                      int* __restrict__ idx)
{
  int q = blockIdx.x * 64 + threadIdx.x;
  if (q >= nq) return;
  float qx = xyz[q * 3], qy = xyz[q * 3 + 1], qz = xyz[q * 3 + 2];
  float bd[32]; int bi[32];
  for (int s = 0; s < 32; ++s) { bd[s] = 1e30f; bi[s] = 0; }
  for (int j = 0; j < n; ++j) {
    float dx = xyz[j * 3] - qx, dy = xyz[j * 3 + 1] - qy, dz = xyz[j * 3 + 2] - qz;
    float d2 = dx * dx + dy * dy + dz * dz;
    if (d2 < bd[31]) {
      int s = 31;
      while (s > 0 && bd[s - 1] > d2) { bd[s] = bd[s - 1]; bi[s] = bi[s - 1]; --s; }
      bd[s] = d2; bi[s] = j;
    }
  }
  int i0 = bi[0];
  for (int s = 0; s < 32; ++s) idx[q * 32 + s] = (bd[s] <= r2) ? bi[s] : i0;
}

// build grouped feature matrix A[(nq*32)][Kp]: rel-xyz + gathered feats
__global__ void k_group(const float* __restrict__ xyz,
                        const h16* __restrict__ feats, int Cf,
                        const int* __restrict__ idx, int nq,
                        h16* __restrict__ A, long Kp)
{
  long i = (long)blockIdx.x * 256 + threadIdx.x;
  if (i >= (long)nq * 32) return;
  int q = (int)(i >> 5);
  int j = idx[i];
  long base = i * Kp;
  A[base + 0] = (h16)(xyz[j * 3 + 0] - xyz[q * 3 + 0]);
  A[base + 1] = (h16)(xyz[j * 3 + 1] - xyz[q * 3 + 1]);
  A[base + 2] = (h16)(xyz[j * 3 + 2] - xyz[q * 3 + 2]);
  for (int c = 0; c < Cf; ++c) A[base + 3 + c] = feats[(long)j * Cf + c];
}

// max over the 32 samples of each group
__global__ void k_gmax(const h16* __restrict__ Y, long Np, int nq, int C,
                       h16* __restrict__ out)
{
  long i = (long)blockIdx.x * 256 + threadIdx.x;
  if (i >= (long)nq * C) return;
  int q = (int)(i / C), c = (int)(i % C);
  float best = -1e30f;
  for (int s = 0; s < 32; ++s)
    best = fmaxf(best, (float)Y[((long)q * 32 + s) * Np + c]);
  out[i] = (h16)best;
}

// 3-NN inverse-distance interpolation of level features
__global__ void k_interp3(const float* __restrict__ q_xyz, int nq,
                          const float* __restrict__ s_xyz, int n2,
                          const h16* __restrict__ f2, int C,
                          h16* __restrict__ out, long Kp)
{
  int q = blockIdx.x * 64 + threadIdx.x;
  if (q >= nq) return;
  float qx = q_xyz[q * 3], qy = q_xyz[q * 3 + 1], qz = q_xyz[q * 3 + 2];
  float d0 = 1e30f, d1 = 1e30f, d2b = 1e30f;
  int i0 = 0, i1 = 0, i2 = 0;
  for (int j = 0; j < n2; ++j) {
    float dx = s_xyz[j * 3] - qx, dy = s_xyz[j * 3 + 1] - qy,
          dz = s_xyz[j * 3 + 2] - qz;
    float d = dx * dx + dy * dy + dz * dz;
    if (d < d0)      { d2b = d1; i2 = i1; d1 = d0; i1 = i0; d0 = d; i0 = j; }
    else if (d < d1) { d2b = d1; i2 = i1; d1 = d; i1 = j; }
    else if (d < d2b){ d2b = d; i2 = j; }
  }
  float w0 = 1.f / fmaxf(d0, 1e-10f);
  float w1 = 1.f / fmaxf(d1, 1e-10f);
  float w2 = 1.f / fmaxf(d2b, 1e-10f);
  float ws = w0 + w1 + w2;
  w0 /= ws; w1 /= ws; w2 /= ws;
  for (int c = 0; c < C; ++c)
    out[(long)q * Kp + c] =
        (h16)(w0 * (float)f2[(long)i0 * C + c] + w1 * (float)f2[(long)i1 * C + c] +
              w2 * (float)f2[(long)i2 * C + c]);
}

__global__ void k_cols_f32(h16* __restrict__ X, long ldx, long col0,
                           const float* __restrict__ src, int n, int c)
{
  long i = (long)blockIdx.x * 256 + threadIdx.x;
  if (i >= (long)n * c) return;
  int p = (int)(i / c), j = (int)(i % c);
  X[(long)p * ldx + col0 + j] = (h16)src[i];
}

__global__ void k_cols_f16(h16* __restrict__ X, long ldx, long col0,
                           const h16* __restrict__ src, long lds, int n, int c)
{
  long i = (long)blockIdx.x * 256 + threadIdx.x;
  if (i >= (long)n * c) return;
  int p = (int)(i / c), j = (int)(i % c);
  X[(long)p * ldx + col0 + j] = src[(long)p * lds + j];
}

__global__ void k_rowpack(const float* __restrict__ v, long n, h16* __restrict__ A)
{
  long i = (long)blockIdx.x * 256 + threadIdx.x;
  if (i >= n) return;
  A[i] = (h16)v[i];
}

__global__ void k_f4r(const h16* __restrict__ fc6o, const float* __restrict__ e5,
                      float* __restrict__ f4r)
{
  long i = (long)blockIdx.x * 256 + threadIdx.x;
  if (i >= 32768) return;
  f4r[i] = (float)fc6o[i] + e5[i];  // relu already applied in GEMM epilogue
}

__global__ void k_final(const h16* __restrict__ Y, long Np,
                        const float* __restrict__ pts, float* __restrict__ out)
{
  long i = (long)blockIdx.x * 256 + threadIdx.x;
  if (i >= 16384L * 3) return;
  int pt = (int)(i / 3), j = (int)(i % 3);
  int p4 = pt >> 2, r = pt & 3;
  out[i] = pts[p4 * 3 + j] + (float)Y[(long)p4 * Np + r * 3 + j];
}

// ------------------------------- host side ---------------------------------

static inline long padc(long x) { return (x + 63) / 64 * 64; }     // N / K pad
static inline long padm(long x) { return (x + 127) / 128 * 128; }  // M pad
static inline unsigned gs(long n, int b) { return (unsigned)((n + b - 1) / b); }

static void gemm(const h16* A, const h16* Bt, const float* bias, h16* C,
                 long M, long N, long K, int act, hipStream_t stream)
{
  long Mp = padm(M), Np = padc(N), Kp = padc(K);
  dim3 g((unsigned)(Np / 64), (unsigned)(Mp / 128));
  k_wmma_gemm<<<g, 256, 0, stream>>>(A, Bt, bias, C, N, Kp, Np, act);
}

static void pack_lin(const float* W, h16* Bt, long K, long N, hipStream_t s)
{
  long Kp = padc(K), Np = padc(N);
  k_pack_lin<<<gs(Kp * Np, 256), 256, 0, s>>>(W, Bt, K, N, Kp, Np);
}

extern "C" void kernel_launch(void* const* d_in, const int* in_sizes, int n_in,
                              void* d_out, int out_size, void* d_ws,
                              size_t ws_size, hipStream_t stream)
{
  (void)in_sizes; (void)n_in; (void)out_size; (void)ws_size;
  const float* points = (const float*)d_in[0];

  // input index map (flattened setup_inputs order):
  // 0 points | 1..10 c{1..5}w,c{1..5}b | 11..15 d{1..5}w | 16,17 fc5 | 18,19 fc6
  // 20..31 fc11..fc16 (w,b) | 32..55 sa[k][j] (w,b) | 56..61 fp[k] (w,b)

  char* base = (char*)d_ws;
  size_t off = 0;
  auto take = [&](size_t bytes) -> void* {
    size_t o = (off + 255) & ~(size_t)255;
    off = o + bytes;
    return (void*)(base + o);
  };

  float* grid = (float*)take(2097152L * 4);      // 128^3
  float* e1 = (float*)take(8388608L * 4);        // 32 x 64^3
  float* e2 = (float*)take(2097152L * 4);        // 64 x 32^3
  float* e3 = (float*)take(524288L * 4);         // 128 x 16^3
  float* e4 = (float*)take(131072L * 4);         // 256 x 8^3
  float* e5 = (float*)take(32768L * 4);          // 512 x 4^3
  float* f4r = (float*)take(32768L * 4);
  float* r1 = (float*)take(131072L * 4);         // 256 x 8^3
  float* r2 = (float*)take(524288L * 4);         // 128 x 16^3
  float* r3 = (float*)take(2097152L * 4);        // 64 x 32^3
  float* r4 = (float*)take(8388608L * 4);        // 32 x 64^3
  float* r5 = (float*)take(2097152L * 4);        // 1 x 128^3
  float* sparse = (float*)take(6291456L * 4);    // 128^3 x 3
  float* stats = (float*)take(1024 * 4);
  int* idxb = (int*)take(4096L * 32 * 4);
  h16* feat1 = (h16*)take(4096L * 64 * 2);
  h16* feat2 = (h16*)take(2048L * 128 * 2);
  h16* feat3 = (h16*)take(1024L * 256 * 2);
  h16* feat4 = (h16*)take(512L * 512 * 2);
  h16* X0 = (h16*)take(4096L * 5504 * 2);
  h16* X1 = (h16*)take(4096L * 5504 * 2);
  h16* SC_A = (h16*)take(16777216L * 2);    // transient A matrices
  h16* SC_W = (h16*)take(134217728L * 2);   // transient packed weights
  h16* SC_O = (h16*)take(137396224L * 2);   // transient GEMM/conv outputs

  // -------- pu feature matrix X0 (columns filled by several stages) --------
  hipMemsetAsync(X0, 0, 4096L * 5504 * 2, stream);
  k_cols_f32<<<gs(4096L * 3, 256), 256, 0, stream>>>(X0, 5504, 0, points, 4096, 3);

  // ------------------------------ SA modules -------------------------------
  {
    const int sa_np[4] = {4096, 2048, 1024, 512};
    const int sa_ns[4] = {4096, 4096, 2048, 1024};
    const float sa_r[4] = {0.05f, 0.1f, 0.2f, 0.3f};
    const int ch[4][4] = {{3, 32, 32, 64}, {67, 64, 64, 128},
                          {131, 128, 128, 256}, {259, 256, 256, 512}};
    h16* featp[5] = {nullptr, feat1, feat2, feat3, feat4};
    for (int k = 0; k < 4; ++k) {
      int nq = sa_np[k], nsrc = sa_ns[k];
      k_knn<<<gs(nq, 64), 64, 0, stream>>>(points, nsrc, nq, sa_r[k] * sa_r[k], idxb);
      int Cf = (k == 0) ? 0 : ch[k][0] - 3;
      long rows = (long)nq * 32, Kp0 = padc(ch[k][0]);
      hipMemsetAsync(SC_A, 0, padm(rows) * Kp0 * 2, stream);
      k_group<<<gs(rows, 256), 256, 0, stream>>>(points, featp[k], Cf, idxb, nq, SC_A, Kp0);
      const h16* Ain = SC_A;
      h16* outs[3] = {SC_O, SC_A, SC_O};
      int Kc = ch[k][0];
      for (int j = 0; j < 3; ++j) {
        int Nc = ch[k][j + 1];
        pack_lin((const float*)d_in[32 + k * 6 + j * 2], SC_W, Kc, Nc, stream);
        gemm(Ain, SC_W, (const float*)d_in[33 + k * 6 + j * 2], outs[j], rows, Nc, Kc, 1, stream);
        Ain = outs[j];
        Kc = Nc;
      }
      k_gmax<<<gs((long)nq * ch[k][3], 256), 256, 0, stream>>>(SC_O, padc(ch[k][3]), nq, ch[k][3], featp[k + 1]);
    }
    k_cols_f16<<<gs(4096L * 64, 256), 256, 0, stream>>>(X0, 5504, 3, feat1, 64, 4096, 64);

    // --------------------------- FP modules -------------------------------
    const int fp_n2[3] = {2048, 1024, 512};
    const int fp_c[3] = {128, 256, 512};
    for (int k = 0; k < 3; ++k) {
      long Kp = padc(fp_c[k]);
      hipMemsetAsync(SC_A, 0, 4096L * Kp * 2, stream);
      k_interp3<<<gs(4096, 64), 64, 0, stream>>>(points, 4096, points, fp_n2[k], featp[k + 2], fp_c[k], SC_A, Kp);
      pack_lin((const float*)d_in[56 + 2 * k], SC_W, fp_c[k], 512, stream);
      gemm(SC_A, SC_W, (const float*)d_in[57 + 2 * k], SC_O, 4096, 512, fp_c[k], 1, stream);
      k_cols_f16<<<gs(4096L * 512, 256), 256, 0, stream>>>(X0, 5504, 67 + 512L * k, SC_O, 512, 4096, 512);
    }
  }

  // ------------------------------ gridding ---------------------------------
  hipMemsetAsync(grid, 0, 2097152L * 4, stream);
  k_gridding<<<gs(4096, 256), 256, 0, stream>>>(points, 4096, grid);

  // ------------------------------- encoder ---------------------------------
  {
    const int din[5] = {128, 64, 32, 16, 8}, dout[5] = {129, 65, 33, 17, 9};
    const int dp[5] = {64, 32, 16, 8, 4};
    const int ci[5] = {1, 32, 64, 128, 256}, co[5] = {32, 64, 128, 256, 512};
    const int widx[5] = {1, 3, 5, 7, 9};
    const float* vin[5] = {grid, e1, e2, e3, e4};
    float* vout[5] = {e1, e2, e3, e4, e5};
    for (int k = 0; k < 5; ++k) {
      long Kp = padc((long)ci[k] * 64), Np = padc(co[k]);
      k_pack_convw<<<gs(Kp * Np, 256), 256, 0, stream>>>((const float*)d_in[widx[k]], SC_W, co[k], ci[k], Kp, Np);
      long M = (long)dout[k] * dout[k] * dout[k], Mp = padm(M);
      dim3 g((unsigned)(Np / 64), (unsigned)(Mp / 128));
      k_wmma_conv<<<g, 256, 0, stream>>>(vin[k], SC_W, SC_O, din[k], dout[k], ci[k], Kp, Np, 0);
      hipMemsetAsync(stats, 0, 4096, stream);
      k_bn_stats<<<gs(M * co[k], 256), 256, 0, stream>>>(SC_O, M, Np, co[k], stats);
      long tot = (long)co[k] * dp[k] * dp[k] * dp[k];
      k_bn_lrelu_pool<<<gs(tot, 256), 256, 0, stream>>>(SC_O, stats, Np, co[k], dout[k], dp[k], vout[k]);
    }
  }

  // ------------------------------ fc5 / fc6 --------------------------------
  hipMemsetAsync(SC_A, 0, 128L * 32768 * 2, stream);
  k_rowpack<<<gs(32768, 256), 256, 0, stream>>>(e5, 32768, SC_A);
  pack_lin((const float*)d_in[16], SC_W, 32768, 4096, stream);
  gemm(SC_A, SC_W, (const float*)d_in[17], SC_O, 1, 4096, 32768, 1, stream);
  pack_lin((const float*)d_in[18], SC_W, 4096, 32768, stream);
  gemm(SC_O, SC_W, (const float*)d_in[19], SC_A, 1, 32768, 4096, 1, stream);
  k_f4r<<<gs(32768, 256), 256, 0, stream>>>(SC_A, e5, f4r);

  // ------------------------------- decoder ---------------------------------
  {
    const int din[5] = {4, 8, 16, 32, 64};
    const int ci[5] = {512, 256, 128, 64, 32}, co[5] = {256, 128, 64, 32, 1};
    const int widx[5] = {11, 12, 13, 14, 15};
    const float* vin[5] = {f4r, r1, r2, r3, r4};
    float* vout[5] = {r1, r2, r3, r4, r5};
    const float* vskip[5] = {e4, e3, e2, e1, grid};
    for (int k = 0; k < 5; ++k) {
      int Dout = 2 * din[k];
      long Kp = padc((long)ci[k] * 64), Np = padc(co[k]);
      k_pack_deconvw<<<gs(Kp * Np, 256), 256, 0, stream>>>((const float*)d_in[widx[k]], SC_W, co[k], ci[k], Kp, Np);
      long M = (long)Dout * Dout * Dout, Mp = padm(M);
      dim3 g((unsigned)(Np / 64), (unsigned)(Mp / 128));
      k_wmma_conv<<<g, 256, 0, stream>>>(vin[k], SC_W, SC_O, din[k], Dout, ci[k], Kp, Np, 1);
      hipMemsetAsync(stats, 0, 4096, stream);
      k_bn_stats<<<gs(M * co[k], 256), 256, 0, stream>>>(SC_O, M, Np, co[k], stats);
      k_bn_relu_add<<<gs((long)co[k] * M, 256), 256, 0, stream>>>(SC_O, stats, Np, co[k], Dout, vskip[k], vout[k]);
    }
  }

  // gridding-reverse (sparse cloud; sampling approximated by partial below)
  k_grid_rev<<<gs(2097152, 256), 256, 0, stream>>>(r5, sparse);

  // ----------------------- cubic feature sampling --------------------------
  k_cubic<<<gs(4096L * 32, 256), 256, 0, stream>>>(points, 4096, r4, 32, 64, X0, 5504, 1603);
  k_cubic<<<gs(4096L * 64, 256), 256, 0, stream>>>(points, 4096, r3, 64, 32, X0, 5504, 1859);
  k_cubic<<<gs(4096L * 128, 256), 256, 0, stream>>>(points, 4096, r2, 128, 16, X0, 5504, 2371);
  k_cubic<<<gs(4096L * 256, 256), 256, 0, stream>>>(points, 4096, r1, 256, 8, X0, 5504, 3395);

  // ------------------------------ FC head ----------------------------------
  {
    const int fw[6] = {20, 22, 24, 26, 28, 30}, fb[6] = {21, 23, 25, 27, 29, 31};
    const int fk[6] = {5443, 5443, 2560, 1280, 640, 240};
    const int fn[6] = {5443, 2560, 1280, 640, 240, 12};
    h16* cur = X0; h16* nxt = X1;
    for (int j = 0; j < 6; ++j) {
      pack_lin((const float*)d_in[fw[j]], SC_W, fk[j], fn[j], stream);
      gemm(cur, SC_W, (const float*)d_in[fb[j]], nxt, 4096, fn[j], fk[j], (j < 5) ? 1 : 0, stream);
      h16* t = cur; cur = nxt; nxt = t;
    }
    k_final<<<gs(16384L * 3, 256), 256, 0, stream>>>(cur, 64, points, (float*)d_out);
  }
}
